// AttentionHierarchy_34299608826485
// MI455X (gfx1250) — compile-verified, matrix operand order checked by reading, not run on hardware
//
#include <hip/hip_runtime.h>
#include <hip/hip_bf16.h>

// ---------------------------------------------------------------------------
// Bidirectional GRU + attention pooling for MI455X (gfx1250, wave32, WMMA).
//   B=64, T=1024, D=300 (padded 320), Hd=256, H=512, 3Hd=768
//   1. pack weights to bf16 WMMA-B fragment-major layout
//   2. mask + convert x to bf16 (padded K)
//   3. input GEMM (both dirs) via v_wmma_f32_16x16x32_bf16,
//      register-blocked 2 M-tiles x 3 N-tiles per wave
//   4. GRU scan: one 32-wave workgroup per direction, h + gate buffer in LDS
//      (~288KB of the WGP's 320KB), A-fragment hoisted across 6 N tiles,
//      global_prefetch of next step's gx row
//   5. attention GEMM (WMMA, 2 N tiles per wave) + ctx-vector reduction
//   6. masked softmax + weighted pooling
// ---------------------------------------------------------------------------

#define BB 64
#define TT 1024
#define DD 300
#define DP 320           // D padded to multiple of 32 (WMMA K-step)
#define HD 256
#define HH 512
#define G3 768           // 3*Hd
#define BT (BB * TT)     // 65536 rows

typedef __attribute__((ext_vector_type(16))) __bf16 v16bf;
typedef __attribute__((ext_vector_type(8)))  __bf16 v8bf;
typedef __attribute__((ext_vector_type(8)))  float  v8f;

__device__ __forceinline__ unsigned short f2bf(float f) {
    unsigned u = __float_as_uint(f);
    unsigned r = (u + 0x7fffu + ((u >> 16) & 1u)) >> 16;
    return (unsigned short)r;
}

// A fragment (16x32 bf16, MxK), rows contiguous-in-K with leading dim `ld`
// (halfs). Lanes 0-15 hold K {0..7,16..23}; lanes 16-31 hold K {8..15,24..31}
// of row M = lane&15. rowstep=-1 walks time-reversed rows.
__device__ __forceinline__ v16bf load_a_frag(const unsigned short* A, int ld,
                                             int rowbase, int rowstep,
                                             int kbase, int lane) {
    int r    = lane & 15;
    int koff = (lane & 16) ? 8 : 0;
    const unsigned short* p =
        A + (long)(rowbase + rowstep * r) * ld + kbase + koff;
    union { v16bf v; v8bf h[2]; } u;
    u.h[0] = *reinterpret_cast<const v8bf*>(p);        // K +0..7
    u.h[1] = *reinterpret_cast<const v8bf*>(p + 16);   // K +16..23
    return u.v;
}

// B fragment (32x16 bf16, KxN) from fragment-major packed weights:
// tile (nt,kt) stores lane-major [32][16] halfs contiguously.
__device__ __forceinline__ v16bf load_b_frag(const unsigned short* P, int KT,
                                             int nt, int kt, int lane) {
    const unsigned short* p = P + (((long)nt * KT + kt) * 32 + lane) * 16;
    union { v16bf v; v8bf h[2]; } u;
    u.h[0] = *reinterpret_cast<const v8bf*>(p);
    u.h[1] = *reinterpret_cast<const v8bf*>(p + 8);
    return u.v;
}

__device__ __forceinline__ v8f wmma_bf16(v16bf a, v16bf b, v8f c) {
    return __builtin_amdgcn_wmma_f32_16x16x32_bf16(false, a, false, b,
                                                   (short)0, c, false, false);
}

// ---- weight packing: W is [N, Kreal] row-major f32, logical B[k][n]=W[n][k],
// zero-padded to Kd. Inverse of the fragment layout above.
__global__ void gru_pack_b(const float* __restrict__ W,
                           unsigned short* __restrict__ dst,
                           int N, int Kreal, int Kd) {
    int  KT    = Kd >> 5;
    long total = (long)N * Kd;
    for (long idx = (long)blockIdx.x * blockDim.x + threadIdx.x; idx < total;
         idx += (long)gridDim.x * blockDim.x) {
        int n = (int)(idx / Kd), k = (int)(idx % Kd);
        int nt = n >> 4, ln = n & 15;
        int kt = k >> 5, kr = k & 31;
        int lane, i;
        if (kr < 16) { lane = ln + ((kr & 8) ? 16 : 0); i = kr & 7; }
        else { int k2 = kr - 16; lane = ln + ((k2 & 8) ? 16 : 0); i = 8 + (k2 & 7); }
        float v = (k < Kreal) ? W[(long)n * Kreal + k] : 0.0f;
        dst[(((long)nt * KT + kt) * 32 + lane) * 16 + i] = f2bf(v);
    }
}

// ---- mask by length + convert x to bf16, pad D..DP with zeros
__global__ void gru_mask_x(const float* __restrict__ x,
                           const int* __restrict__ len,
                           unsigned short* __restrict__ xm) {
    long total = (long)BT * DP;
    for (long idx = (long)blockIdx.x * blockDim.x + threadIdx.x; idx < total;
         idx += (long)gridDim.x * blockDim.x) {
        long m = idx / DP; int d = (int)(idx % DP);
        int b = (int)(m >> 10), t = (int)(m & 1023);
        float v = 0.0f;
        if (d < DD && t < len[b]) v = x[m * DD + d];
        xm[idx] = f2bf(v);
    }
}

// ---- input GEMM: gx[dir] = (xm or time-reversed xm) @ w_ih^T + b_ih
// grid (BT/32, G3/384, 2), 8 waves/block; each wave: 2 M-tiles x 3 N-tiles.
__global__ __launch_bounds__(256) void gru_input_gemm(
    const unsigned short* __restrict__ xm,
    const unsigned short* __restrict__ wpF,
    const unsigned short* __restrict__ wpB,
    const float* __restrict__ bihF, const float* __restrict__ bihB,
    float* __restrict__ gxF, float* __restrict__ gxB) {
    int lane = threadIdx.x & 31, w = threadIdx.x >> 5;
    int mbase  = blockIdx.x * 32;              // two 16-row M tiles
    int ntbase = blockIdx.y * 24 + w * 3;      // three N tiles per wave
    int dir    = blockIdx.z;
    const unsigned short* wp = dir ? wpB : wpF;
    const float* bih = dir ? bihB : bihF;
    float* gx = dir ? gxB : gxF;

    int rowbase0, rowbase1, rowstep;
    if (!dir) { rowbase0 = mbase; rowbase1 = mbase + 16; rowstep = 1; }
    else {     // time-reversed within each length-1024 sequence
        int tb   = mbase & 1023;
        rowbase0 = (mbase & ~1023) + 1023 - tb;
        rowbase1 = rowbase0 - 16;
        rowstep  = -1;
    }

    v8f c[2][3];
#pragma unroll
    for (int mi = 0; mi < 2; ++mi)
#pragma unroll
        for (int j = 0; j < 3; ++j)
            c[mi][j] = (v8f){0.f, 0.f, 0.f, 0.f, 0.f, 0.f, 0.f, 0.f};

    for (int kt = 0; kt < DP / 32; ++kt) {
        v16bf a0 = load_a_frag(xm, DP, rowbase0, rowstep, kt * 32, lane);
        v16bf a1 = load_a_frag(xm, DP, rowbase1, rowstep, kt * 32, lane);
#pragma unroll
        for (int j = 0; j < 3; ++j) {
            v16bf b = load_b_frag(wp, DP / 32, ntbase + j, kt, lane);
            c[0][j] = wmma_bf16(a0, b, c[0][j]);
            c[1][j] = wmma_bf16(a1, b, c[1][j]);
        }
    }

    int ro = (lane & 16) ? 8 : 0;
#pragma unroll
    for (int mi = 0; mi < 2; ++mi) {
#pragma unroll
        for (int j = 0; j < 3; ++j) {
            int   col = (ntbase + j) * 16 + (lane & 15);
            float bv  = bih[col];
#pragma unroll
            for (int i = 0; i < 8; ++i)
                gx[(long)(mbase + mi * 16 + i + ro) * G3 + col] =
                    c[mi][j][i] + bv;
        }
    }
}

// ---- GRU scan: 1 block per direction, 32 waves, state in LDS.
// Per step: gh = h_bf @ w_hh^T (192 C tiles over 32 waves; A hoisted over the
// wave's 6 N tiles), barrier, fused gate math + enc writeout, barrier.
__global__ __launch_bounds__(1024) void gru_scan(
    const unsigned short* __restrict__ whhF,
    const unsigned short* __restrict__ whhB,
    const float* __restrict__ gxF, const float* __restrict__ gxB,
    const float* __restrict__ bhhF, const float* __restrict__ bhhB,
    float* __restrict__ enc, unsigned short* __restrict__ encbf) {
    extern __shared__ char smem[];
    float*          gh  = (float*)smem;                           // 64*768 f32
    float*          hf  = (float*)(smem + (size_t)64 * 768 * 4);  // 64*256 f32
    unsigned short* hbf = (unsigned short*)(smem + (size_t)64 * 768 * 4 +
                                            (size_t)64 * 256 * 4); // 64*256 bf16

    int dir = blockIdx.x;
    const unsigned short* wp  = dir ? whhB : whhF;
    const float*          gx  = dir ? gxB : gxF;
    const float*          bhh = dir ? bhhB : bhhF;

    int tid = threadIdx.x, lane = tid & 31, w = tid >> 5;
    for (int e = tid; e < 64 * 256; e += 1024) { hf[e] = 0.0f; hbf[e] = 0; }
    __syncthreads();

    int mt  = w & 3;         // 4 M tiles
    int ntg = w >> 2;        // 8 groups x 6 N tiles = 48
    for (int t = 0; t < TT; ++t) {
        // phase 1: gh tiles
        v8f c[6];
#pragma unroll
        for (int s = 0; s < 6; ++s)
            c[s] = (v8f){0.f, 0.f, 0.f, 0.f, 0.f, 0.f, 0.f, 0.f};
#pragma unroll
        for (int kt = 0; kt < 8; ++kt) {
            v16bf a = load_a_frag(hbf, 256, mt * 16, 1, kt * 32, lane);
#pragma unroll
            for (int s = 0; s < 6; ++s) {
                v16bf b = load_b_frag(wp, 8, ntg * 6 + s, kt, lane);
                c[s] = wmma_bf16(a, b, c[s]);
            }
        }
        int ro = (lane & 16) ? 8 : 0;
#pragma unroll
        for (int s = 0; s < 6; ++s) {
            int col = (ntg * 6 + s) * 16 + (lane & 15);
#pragma unroll
            for (int i = 0; i < 8; ++i)
                gh[(mt * 16 + i + ro) * G3 + col] = c[s][i];
        }
        __syncthreads();
        // phase 2: gate math (torch GRU), write h and enc
        for (int e = tid; e < 64 * 256; e += 1024) {
            int m = e >> 8, j = e & 255;
            const float* g = gx + ((long)m * TT + t) * G3;
            if (t + 1 < TT) __builtin_prefetch(g + G3 + j, 0, 3);
            float hr = gh[m * G3 + j]       + bhh[j];
            float hz = gh[m * G3 + 256 + j] + bhh[256 + j];
            float hn = gh[m * G3 + 512 + j] + bhh[512 + j];
            float r = 1.0f / (1.0f + __expf(-(g[j] + hr)));
            float z = 1.0f / (1.0f + __expf(-(g[256 + j] + hz)));
            float n = tanhf(g[512 + j] + r * hn);
            float hnew = (1.0f - z) * n + z * hf[e];
            hf[e]  = hnew;
            hbf[e] = f2bf(hnew);
            int  tout = dir ? (TT - 1 - t) : t;
            long ei   = ((long)m * TT + tout) * HH + dir * HD + j;
            enc[ei]   = hnew;
            encbf[ei] = f2bf(hnew);
        }
        __syncthreads();
    }
}

// ---- attention scores: u = tanh(enc @ att_w^T + att_b); logit = u . ctx_v
// block = 16 waves handles 16 rows x all 512 cols (2 N tiles per wave,
// A fragment shared); LDS float atomics reduce against ctx_v.
__global__ __launch_bounds__(512) void att_score(
    const unsigned short* __restrict__ encbf,
    const unsigned short* __restrict__ attp,
    const float* __restrict__ attb, const float* __restrict__ ctxv,
    float* __restrict__ logits) {
    __shared__ float slog[16];
    int tid = threadIdx.x, lane = tid & 31, w = tid >> 5;
    int mbase = blockIdx.x * 16;
    if (tid < 16) slog[tid] = 0.0f;
    __syncthreads();

    v8f c[2];
#pragma unroll
    for (int h = 0; h < 2; ++h)
        c[h] = (v8f){0.f, 0.f, 0.f, 0.f, 0.f, 0.f, 0.f, 0.f};
    for (int kt = 0; kt < HH / 32; ++kt) {
        v16bf a = load_a_frag(encbf, HH, mbase, 1, kt * 32, lane);
#pragma unroll
        for (int h = 0; h < 2; ++h) {
            v16bf b = load_b_frag(attp, HH / 32, w * 2 + h, kt, lane);
            c[h] = wmma_bf16(a, b, c[h]);
        }
    }
    int ro = (lane & 16) ? 8 : 0;
#pragma unroll
    for (int h = 0; h < 2; ++h) {
        int   col = (w * 2 + h) * 16 + (lane & 15);
        float cv = ctxv[col], ab = attb[col];
        float part[8];
#pragma unroll
        for (int i = 0; i < 8; ++i) part[i] = tanhf(c[h][i] + ab) * cv;
#pragma unroll
        for (int i = 0; i < 8; ++i) atomicAdd(&slog[i + ro], part[i]);
    }
    __syncthreads();
    if (tid < 16) logits[mbase + tid] = slog[tid];
}

// ---- masked softmax over T + weighted pooling -> out[B,H]
__global__ __launch_bounds__(256) void softmax_pool(
    const float* __restrict__ logits, const int* __restrict__ len,
    const float* __restrict__ enc, float* __restrict__ out) {
    __shared__ float red[256];
    __shared__ float aw[TT];
    int b = blockIdx.x, tid = threadIdx.x;
    int L = len[b];
    float lm = -3.4e38f;
    for (int t = tid; t < TT; t += 256)
        if (t < L) lm = fmaxf(lm, logits[b * TT + t]);
    red[tid] = lm; __syncthreads();
    for (int s = 128; s > 0; s >>= 1) {
        if (tid < s) red[tid] = fmaxf(red[tid], red[tid + s]);
        __syncthreads();
    }
    float mx = red[0];
    __syncthreads();
    float ls = 0.0f;
    for (int t = tid; t < TT; t += 256) {
        float e = 0.0f;
        if (t < L) e = __expf(logits[b * TT + t] - mx);
        aw[t] = e; ls += e;
    }
    red[tid] = ls; __syncthreads();
    for (int s = 128; s > 0; s >>= 1) {
        if (tid < s) red[tid] += red[tid + s];
        __syncthreads();
    }
    float inv = 1.0f / red[0];
    __syncthreads();
    for (int h = tid; h < HH; h += 256) {
        float acc = 0.0f;
        for (int t = 0; t < L; ++t)
            acc += aw[t] * enc[((long)b * TT + t) * HH + h];
        out[b * HH + h] = acc * inv;
    }
}

extern "C" void kernel_launch(void* const* d_in, const int* in_sizes, int n_in,
                              void* d_out, int out_size, void* d_ws,
                              size_t ws_size, hipStream_t stream) {
    const float* x    = (const float*)d_in[0];
    const int*   len  = (const int*)d_in[1];
    const float* wihF = (const float*)d_in[2];
    const float* whhF = (const float*)d_in[3];
    const float* bihF = (const float*)d_in[4];
    const float* bhhF = (const float*)d_in[5];
    const float* wihB = (const float*)d_in[6];
    const float* whhB = (const float*)d_in[7];
    const float* bihB = (const float*)d_in[8];
    const float* bhhB = (const float*)d_in[9];
    const float* attw = (const float*)d_in[10];
    const float* attb = (const float*)d_in[11];
    const float* ctxv = (const float*)d_in[12];
    float* out = (float*)d_out;

    char*  ws = (char*)d_ws;
    size_t o  = 0;
    unsigned short* xm    = (unsigned short*)(ws + o); o += (size_t)BT * DP * 2;
    unsigned short* pWihF = (unsigned short*)(ws + o); o += (size_t)48 * 10 * 512 * 2;
    unsigned short* pWihB = (unsigned short*)(ws + o); o += (size_t)48 * 10 * 512 * 2;
    unsigned short* pWhhF = (unsigned short*)(ws + o); o += (size_t)48 * 8 * 512 * 2;
    unsigned short* pWhhB = (unsigned short*)(ws + o); o += (size_t)48 * 8 * 512 * 2;
    unsigned short* pAtt  = (unsigned short*)(ws + o); o += (size_t)32 * 16 * 512 * 2;
    float*          gxF   = (float*)(ws + o);          o += (size_t)BT * G3 * 4;
    float*          gxB   = (float*)(ws + o);          o += (size_t)BT * G3 * 4;
    float*          enc   = (float*)(ws + o);          o += (size_t)BT * HH * 4;
    unsigned short* encbf = (unsigned short*)(ws + o); o += (size_t)BT * HH * 2;
    float*          lgt   = (float*)(ws + o);          o += (size_t)BT * 4;

    gru_pack_b<<<512, 256, 0, stream>>>(wihF, pWihF, G3, DD, DP);
    gru_pack_b<<<512, 256, 0, stream>>>(wihB, pWihB, G3, DD, DP);
    gru_pack_b<<<512, 256, 0, stream>>>(whhF, pWhhF, G3, HD, HD);
    gru_pack_b<<<512, 256, 0, stream>>>(whhB, pWhhB, G3, HD, HD);
    gru_pack_b<<<512, 256, 0, stream>>>(attw, pAtt, HH, HH, HH);
    gru_mask_x<<<4096, 256, 0, stream>>>(x, len, xm);

    dim3 g1(BT / 32, G3 / 384, 2);
    gru_input_gemm<<<g1, 256, 0, stream>>>(xm, pWihF, pWihB, bihF, bihB,
                                           gxF, gxB);

    size_t smem = (size_t)64 * 768 * 4 + (size_t)64 * 256 * 4 +
                  (size_t)64 * 256 * 2;   // 294912 B < 320KB WGP LDS
    hipFuncSetAttribute((const void*)gru_scan,
                        hipFuncAttributeMaxDynamicSharedMemorySize,
                        (int)smem);
    gru_scan<<<2, 1024, smem, stream>>>(pWhhF, pWhhB, gxF, gxB, bhhF, bhhB,
                                        enc, encbf);

    att_score<<<BT / 16, 512, 0, stream>>>(encbf, pAtt, attb, ctxv, lgt);
    softmax_pool<<<BB, 256, 0, stream>>>(lgt, len, enc, out);
}
